// HyperbolicFullAttention_87832081203969
// MI455X (gfx1250) — matrix operand order repaired and checked by reading, not compiled, not run on hardware
//
#include <hip/hip_runtime.h>

typedef __attribute__((ext_vector_type(16))) _Float16 v16h;
typedef __attribute__((ext_vector_type(8)))  _Float16 v8h;
typedef __attribute__((ext_vector_type(8)))  float    v8f;

#define Bb   4
#define Nn   2048
#define KIN  129      // D_IN + 1
#define Hh   8
#define Dq   64       // D_OUT
#define Aa   65       // ambient dim D_OUT+1
#define CH   2        // K chunks for QK (64 spatial features, no padding)
#define AP   80       // padded ambient dim for V (5 x 16); feature 65 == ones
#define APAD 68       // per-head mid row stride (floats)
#define WPB  4        // waves per block in attention kernel
#define TQ   2        // q-tiles (16 queries each) per wave
#define NC   (Nn/32)  // key chunks

// ---- swizzled tile index helpers -------------------------------------------
// Q / Kt tiles (spatial only): [b][h][ntile(N/16)][kchunk(2)][lane(32)][elem(16)]
#define QS_IDX(b,h,nt,c,lane,e) \
  ((((((size_t)(b)*Hh + (h))*(Nn/16) + (size_t)(nt))*CH + (size_t)(c))*32 + (size_t)(lane))*16 + (size_t)(e))
// V tiles: [b][h][mchunk(N/32)][atile(5)][lane(32)][elem(16)]
#define VS_IDX(b,h,mc,t,lane,e) \
  ((((((size_t)(b)*Hh + (h))*(Nn/32) + (size_t)(mc))*5 + (size_t)(t))*32 + (size_t)(lane))*16 + (size_t)(e))

// A-matrix (16x32 f16): elems 0..7 -> K=hs*8+e ; elems 8..15 -> K=16+hs*8+(e-8)
__device__ __forceinline__ void qa_map(int kk, int& c, int& hs, int& e) {
  c = kk >> 5; int loc = kk & 31;
  if (loc < 16) { hs = loc >> 3; e = loc & 7; }
  else          { loc -= 16; hs = loc >> 3; e = 8 + (loc & 7); }
}
// B-matrix (32x16 f16): elems 0..7 -> K=hs*8+e ; elems 8..15 -> K=16+hs*8+(e-8)
__device__ __forceinline__ void kb_map(int kk, int& c, int& hs, int& e) {
  c = kk >> 5; int loc = kk & 31; hs = (loc >> 3) & 1; e = (loc & 7) + ((loc >> 4) << 3);
}

// CDNA5 async global->LDS copy (512B per instruction: 32 lanes x 16B), ASYNCcnt.
__device__ __forceinline__ void async_copy_b128(unsigned lds_off, const void* g) {
  asm volatile("global_load_async_to_lds_b128 %0, %1, off"
               :: "v"(lds_off), "v"(g) : "memory");
}

// ============================================================================
// Kernel 1: projections q/k/v = lift(x@W + b), f16, pre-swizzled for WMMA.
// Q/K tiles hold only the 64 spatial features (time handled rank-1 in attn);
// K time is stored negated in nk0g.  V keeps time at feature 0 and a "ones"
// column at feature 65 so the PV WMMA accumulates the softmax denominator.
// ============================================================================
__global__ __launch_bounds__(512) void proj_kernel(
    const float* __restrict__ x,
    const float* __restrict__ Wq, const float* __restrict__ Wk, const float* __restrict__ Wv,
    const float* __restrict__ bq, const float* __restrict__ bk, const float* __restrict__ bv,
    _Float16* __restrict__ qs, _Float16* __restrict__ kt, _Float16* __restrict__ vs,
    float* __restrict__ tqg, float* __restrict__ nk0g)
{
  __shared__ float xs[16][KIN + 3];
  __shared__ float red[Hh][2][16][3];

  const int tid = threadIdx.x;
  const int h = tid >> 6, d = tid & 63;
  const int b = blockIdx.y;
  const int n0 = blockIdx.x * 16;

  for (int idx = tid; idx < 16 * KIN; idx += 512) {
    int r = idx / KIN, i = idx - r * KIN;
    xs[r][i] = x[((size_t)(b * Nn + n0 + r)) * KIN + i];
  }
  __syncthreads();

  float aq[16], ak[16], av[16];
  const float bqv = bq[h * Dq + d], bkv = bk[h * Dq + d], bvv = bv[h * Dq + d];
#pragma unroll
  for (int n = 0; n < 16; ++n) { aq[n] = bqv; ak[n] = bkv; av[n] = bvv; }

  for (int i = 0; i < KIN; ++i) {
    const float wq = Wq[(h * KIN + i) * Dq + d];
    const float wk = Wk[(h * KIN + i) * Dq + d];
    const float wv = Wv[(h * KIN + i) * Dq + d];
#pragma unroll
    for (int n = 0; n < 16; ++n) {
      const float xv = xs[n][i];
      aq[n] = fmaf(xv, wq, aq[n]);
      ak[n] = fmaf(xv, wk, ak[n]);
      av[n] = fmaf(xv, wv, av[n]);
    }
  }

  // ||spatial||^2 per (h,n): reduce across 32 lanes, combine the 2 half-heads.
#pragma unroll
  for (int n = 0; n < 16; ++n) {
    float q2 = aq[n] * aq[n], k2 = ak[n] * ak[n], v2 = av[n] * av[n];
#pragma unroll
    for (int m = 1; m < 32; m <<= 1) {
      q2 += __shfl_xor(q2, m, 32);
      k2 += __shfl_xor(k2, m, 32);
      v2 += __shfl_xor(v2, m, 32);
    }
    if ((d & 31) == 0) {
      red[h][d >> 5][n][0] = q2;
      red[h][d >> 5][n][1] = k2;
      red[h][d >> 5][n][2] = v2;
    }
  }
  __syncthreads();

  for (int n = 0; n < 16; ++n) {
    const int m = n0 + n;
    const int nt16 = m >> 4, lr = m & 15;
    const int mc = m >> 5, hsv = (m >> 4) & 1, ev = m & 15;

    // spatial feature kk = d (0..63)
    { int c, hs, e; qa_map(d, c, hs, e);
      qs[QS_IDX(b, h, nt16, c, hs * 16 + lr, e)] = (_Float16)aq[n]; }
    { int c, hs, e; kb_map(d, c, hs, e);
      kt[QS_IDX(b, h, nt16, c, hs * 16 + lr, e)] = (_Float16)ak[n]; }
    { const int a = d + 1;   // V: time at 0, spatial at 1..64
      vs[VS_IDX(b, h, mc, a >> 4, hsv * 16 + (a & 15), ev)] = (_Float16)av[n]; }

    if (d == 0) {  // time components
      const float tq = sqrtf(1.0f + red[h][0][n][0] + red[h][1][n][0]);
      const float tk = sqrtf(1.0f + red[h][0][n][1] + red[h][1][n][1]);
      const float tv = sqrtf(1.0f + red[h][0][n][2] + red[h][1][n][2]);
      tqg[((size_t)(b * Hh + h)) * Nn + m]  = tq;
      nk0g[((size_t)(b * Hh + h)) * Nn + m] = -tk;   // negated K time
      vs[VS_IDX(b, h, mc, 0, hsv * 16, ev)] = (_Float16)tv;
    }
    if (d < 15) {  // V features 65..79: ones column at 65, zero pads above
      const int a = 65 + d;
      vs[VS_IDX(b, h, mc, a >> 4, hsv * 16 + (a & 15), ev)] =
          (_Float16)((d == 0) ? 1.0f : 0.0f);
    }
  }
}

// ============================================================================
// Kernel 2: flash attention.  Block = 4 waves x 2 q-tiles = 128 queries of one
// (b,h).  K/V chunks (9KB / 32 keys) are staged into LDS ONCE per block with
// async global->LDS copies (double-buffered, ASYNCcnt + split barriers) and
// consumed by all 4 waves.  Softmax uses the hyperboloid bound <q,k>_L <= -1
// (fixed shift, no running max); the denominator rides the PV WMMA through
// V's ones-column.  The -q0*k0 time term is a rank-1 f32 correction.
// ============================================================================
__global__ __launch_bounds__(WPB * 32) void attn_kernel(
    const _Float16* __restrict__ qs, const _Float16* __restrict__ kt,
    const _Float16* __restrict__ vs, const float* __restrict__ tqg,
    const float* __restrict__ nk0g, const float* __restrict__ scale_p,
    float* __restrict__ midg)
{
  // chunk layout in LDS: 8 segments Kt (4KB: ntile {0,1} x chunk {0,1}) then
  // 10 segments V (5KB: atile 0..4); segment = 512B.
  __shared__ alignas(32) _Float16 kvbuf[2][18 * 256];
  __shared__ alignas(32) _Float16 pl[WPB][TQ][16][32];
  __shared__ float nk0s[Nn];

  const int lane = threadIdx.x & 31;
  const int wid  = threadIdx.x >> 5;
  const int hs = lane >> 4, col = lane & 15;
  const int qt0 = (blockIdx.x * WPB + wid) * TQ;
  const int h = blockIdx.y, b = blockIdx.z;

  // -k0 for the whole key row -> LDS (shared by all waves)
  for (int i = threadIdx.x; i < Nn; i += WPB * 32)
    nk0s[i] = nk0g[((size_t)(b * Hh + h)) * Nn + i];
  __syncthreads();

  const float sc  = 2.0f / (scale_p[0] + 1e-7f);
  const float sc2 = sc * 1.44269504088896340736f;   // exp2 domain

  v16h qa[TQ][CH];
  float q0r[TQ][8];
#pragma unroll
  for (int u = 0; u < TQ; ++u) {
#pragma unroll
    for (int c = 0; c < CH; ++c)
      qa[u][c] = *(const v16h*)(qs + QS_IDX(b, h, qt0 + u, c, lane, 0));
#pragma unroll
    for (int r = 0; r < 8; ++r)
      q0r[u][r] = tqg[((size_t)(b * Hh + h)) * Nn + (qt0 + u) * 16 + r + 8 * hs];
  }

  v8f O[TQ][5];
#pragma unroll
  for (int u = 0; u < TQ; ++u)
#pragma unroll
    for (int t = 0; t < 5; ++t) { v8f z = {}; O[u][t] = z; }

  // (b,h) bases for the staged stream (halves)
  const _Float16* ktbh = kt + ((size_t)(b * Hh + h)) * (Nn / 16) * (CH * 512);
  const _Float16* vsbh = vs + ((size_t)(b * Hh + h)) * (Nn / 32) * (5 * 512);
  const unsigned ldsb0 = (unsigned)(size_t)&kvbuf[0][0];
  const unsigned ldsb1 = (unsigned)(size_t)&kvbuf[1][0];

  // each wave issues exactly 5 async copies per chunk (segment 17 duplicated)
  auto issue_chunk = [&](int mc2, unsigned base) {
#pragma unroll
    for (int k = 0; k < 5; ++k) {
      int s = wid + 4 * k; if (s > 17) s = 17;
      const _Float16* g = (s < 8)
          ? (ktbh + (size_t)mc2 * 2048 + s * 256 + lane * 8)
          : (vsbh + (size_t)mc2 * 2560 + (s - 8) * 256 + lane * 8);
      async_copy_b128(base + (unsigned)(s * 512 + lane * 16), g);
    }
  };

  issue_chunk(0, ldsb0);
  unsigned cur = 0;

  for (int mc2 = 0; mc2 < NC; ++mc2) {
    if (mc2 + 1 < NC) {
      issue_chunk(mc2 + 1, cur ? ldsb0 : ldsb1);
      asm volatile("s_wait_asynccnt 5" ::: "memory");  // chunk mc2 copies done
    } else {
      asm volatile("s_wait_asynccnt 0" ::: "memory");
    }
    __syncthreads();                                    // visible to all waves
    const _Float16* kv = cur ? &kvbuf[1][0] : &kvbuf[0][0];

    // ---- S = Qs @ Ks^T (spatial, K=64) -------------------------------------
    v8f s[TQ][2];
#pragma unroll
    for (int u = 0; u < TQ; ++u) { v8f z = {}; s[u][0] = z; s[u][1] = z; }
#pragma unroll
    for (int j = 0; j < 2; ++j)
#pragma unroll
      for (int c = 0; c < CH; ++c) {
        v16h kb = *(const v16h*)(kv + (j * CH + c) * 512 + lane * 16);
#pragma unroll
        for (int u = 0; u < TQ; ++u)
          s[u][j] = __builtin_amdgcn_wmma_f32_16x16x32_f16(
              false, qa[u][c], false, kb, (short)0, s[u][j], false, false);
      }

    // ---- rank-1 time term, p = exp2(sc2*(li+1)) -> LDS ---------------------
    // args are <= 0 (hyperboloid bound); raw v_exp_f32 handles the full range.
    const float nka = nk0s[mc2 * 32 + col];
    const float nkb = nk0s[mc2 * 32 + 16 + col];
#pragma unroll
    for (int u = 0; u < TQ; ++u)
#pragma unroll
      for (int r = 0; r < 8; ++r) {
        const float li0 = fmaf(q0r[u][r], nka, s[u][0][r]);
        const float li1 = fmaf(q0r[u][r], nkb, s[u][1][r]);
        pl[wid][u][r + 8 * hs][col] =
            (_Float16)__builtin_amdgcn_exp2f(fmaf(li0, sc2, sc2));
        pl[wid][u][r + 8 * hs][col + 16] =
            (_Float16)__builtin_amdgcn_exp2f(fmaf(li1, sc2, sc2));
      }

    asm volatile("" ::: "memory");   // intra-wave DS in order; block reordering
    union { v16h v; v8h p[2]; } pa[TQ];
#pragma unroll
    for (int u = 0; u < TQ; ++u) {
      pa[u].p[0] = *(const v8h*)&pl[wid][u][col][hs * 16];
      pa[u].p[1] = *(const v8h*)&pl[wid][u][col][hs * 16 + 8];
    }
    asm volatile("" ::: "memory");

    // ---- O += P @ V (5 feature tiles incl. ones column) --------------------
#pragma unroll
    for (int t = 0; t < 5; ++t) {
      v16h vb = *(const v16h*)(kv + 2048 + t * 512 + lane * 16);
#pragma unroll
      for (int u = 0; u < TQ; ++u)
        O[u][t] = __builtin_amdgcn_wmma_f32_16x16x32_f16(
            false, pa[u].v, false, vb, (short)0, O[u][t], false, false);
    }

    __syncthreads();   // all waves done reading kv before it is overwritten
    cur ^= 1u;
  }

  // ---- epilogue: l = O[4] col 1; mu = O/l; Lorentz normalize; store mids ---
#pragma unroll
  for (int u = 0; u < TQ; ++u) {
    float inv_l[8];
#pragma unroll
    for (int r = 0; r < 8; ++r)
      inv_l[r] = 1.0f / __shfl(O[u][4][r], hs * 16 + 1, 32);

    float mu[5][8], lp[8];
#pragma unroll
    for (int r = 0; r < 8; ++r) lp[r] = 0.0f;
#pragma unroll
    for (int t = 0; t < 5; ++t) {
      const int fidx = t * 16 + col;
      const float w = (fidx == 0) ? -1.0f : (fidx < Aa ? 1.0f : 0.0f);
#pragma unroll
      for (int r = 0; r < 8; ++r) {
        mu[t][r] = O[u][t][r] * inv_l[r];
        lp[r] += w * mu[t][r] * mu[t][r];
      }
    }
    float invn[8];
#pragma unroll
    for (int r = 0; r < 8; ++r) {
      float v = lp[r];
#pragma unroll
      for (int msk = 1; msk < 16; msk <<= 1) v += __shfl_xor(v, msk, 32);
      invn[r] = rsqrtf(fmaxf(-v, 1e-7f));   // sqrt(c)=1, clip(-lor, EPS)
    }
#pragma unroll
    for (int t = 0; t < 5; ++t) {
      const int fidx = t * 16 + col;
      if (fidx < Aa) {
#pragma unroll
        for (int r = 0; r < 8; ++r) {
          const int q = (qt0 + u) * 16 + r + 8 * hs;
          midg[(((size_t)(b * Nn + q)) * Hh + h) * APAD + fidx] = mu[t][r] * invn[r];
        }
      }
    }
  }
}

// ============================================================================
// Kernel 3: head mean -> Lorentz normalize -> re-lift (C_ATTN=C_OUT=1).
// ============================================================================
__global__ __launch_bounds__(256) void merge_kernel(
    const float* __restrict__ midg, float* __restrict__ out)
{
  const int t = blockIdx.x * blockDim.x + threadIdx.x;
  if (t >= Bb * Nn) return;
  const float* base = midg + (size_t)t * Hh * APAD;

  float acc[Aa];
#pragma unroll
  for (int a = 0; a < Aa; ++a) acc[a] = 0.0f;
  for (int h = 0; h < Hh; ++h) {
#pragma unroll
    for (int a = 0; a < Aa; ++a) acc[a] += base[h * APAD + a];
  }
  float lor = 0.0f;
#pragma unroll
  for (int a = 0; a < Aa; ++a) {
    const float v = acc[a] * (1.0f / Hh);
    acc[a] = v;
    lor += (a == 0) ? -v * v : v * v;
  }
  const float inv = rsqrtf(fmaxf(-lor, 1e-7f));
  float ss = 0.0f;
#pragma unroll
  for (int a = 1; a < Aa; ++a) {
    const float s = acc[a] * inv;   // * sqrt(C_ATTN/C_OUT) = 1
    acc[a] = s;
    ss += s * s;
  }
  float* o = out + (size_t)t * Aa;
  o[0] = sqrtf(1.0f + ss);          // re-lift time component
#pragma unroll
  for (int a = 1; a < Aa; ++a) o[a] = acc[a];
}

// ============================================================================
extern "C" void kernel_launch(void* const* d_in, const int* in_sizes, int n_in,
                              void* d_out, int out_size, void* d_ws, size_t ws_size,
                              hipStream_t stream) {
  (void)in_sizes; (void)n_in; (void)out_size; (void)ws_size;
  const float* x     = (const float*)d_in[0];
  const float* Wq    = (const float*)d_in[1];
  const float* Wk    = (const float*)d_in[2];
  const float* Wv    = (const float*)d_in[3];
  const float* bqp   = (const float*)d_in[4];
  const float* bkp   = (const float*)d_in[5];
  const float* bvp   = (const float*)d_in[6];
  const float* scale = (const float*)d_in[7];
  // d_in[8] = attn_bias: row-constant, cancels in softmax.

  char* ws = (char*)d_ws;
  const size_t qs_bytes = (size_t)Bb * Hh * Nn * (CH * 32) * 2;  // 8.39 MB
  const size_t vs_bytes = (size_t)Bb * Hh * Nn * AP * 2;         // 10.5 MB
  const size_t t_bytes  = (size_t)Bb * Hh * Nn * 4;              // 256 KB
  _Float16* qsw = (_Float16*)ws;
  _Float16* ktw = (_Float16*)(ws + qs_bytes);
  _Float16* vsw = (_Float16*)(ws + 2 * qs_bytes);
  float*    tqg = (float*)(ws + 2 * qs_bytes + vs_bytes);
  float*    nk0 = (float*)(ws + 2 * qs_bytes + vs_bytes + t_bytes);
  float*    mid = (float*)(ws + 2 * qs_bytes + vs_bytes + 2 * t_bytes);

  proj_kernel<<<dim3(Nn / 16, Bb), 512, 0, stream>>>(
      x, Wq, Wk, Wv, bqp, bkp, bvp, qsw, ktw, vsw, tqg, nk0);
  attn_kernel<<<dim3(Nn / (16 * TQ * WPB), Hh, Bb), WPB * 32, 0, stream>>>(
      qsw, ktw, vsw, tqg, nk0, scale, mid);
  merge_kernel<<<(Bb * Nn + 255) / 256, 256, 0, stream>>>(mid, (float*)d_out);
}
